// MM_GCN_75746043232444
// MI455X (gfx1250) — matrix operand order repaired and checked by reading, not compiled, not run on hardware
//
#include <hip/hip_runtime.h>
#include <hip/hip_bf16.h>
#include <math.h>

// ---------------------------------------------------------------------------
// MM-GCN for MI455X (gfx1250, wave32, WMMA).
// Block-sparse adjacency never materialized (42 nnz/row). Dense GEMMs
// (12000x256x256, x5) on v_wmma_f32_16x16x32_f16 with f32 accumulation.
// Operands converted to f16 at PRODUCE time; weights pre-swizzled into
// fragment order so B fragments are single contiguous 32B/lane loads and
// A fragments are two ds_load_b128 from a padded LDS panel.
// ---------------------------------------------------------------------------

#define NTOT   4000     // B_DIA * L_DIA
#define BD     100
#define LD     40
#define DIM    256
#define ROWS3  (3 * NTOT)   // 12000
#define WFRAG  (16 * 8 * 32 * 16)   // 65536 f16 per weight matrix

typedef __attribute__((ext_vector_type(16))) _Float16 v16h;
typedef __attribute__((ext_vector_type(8)))  _Float16 v8h;
typedef __attribute__((ext_vector_type(8)))  float    v8f;

// ---------------------------------------------------------------------------
// prep: X = [a ; v ; l + spk_emb[argmax(q)]], X16 = f16(X), NRM = rownorm(X)
// one wave per row, 8 rows per 256-thread block
// ---------------------------------------------------------------------------
__global__ __launch_bounds__(256)
void prep_kernel(const float* __restrict__ a, const float* __restrict__ v,
                 const float* __restrict__ l, const float* __restrict__ qmask,
                 const float* __restrict__ spk_emb,
                 float* __restrict__ X, _Float16* __restrict__ X16,
                 float* __restrict__ NRM)
{
    const int wave = threadIdx.x >> 5;
    const int lane = threadIdx.x & 31;
    const int row  = blockIdx.x * 8 + wave;          // grid = 1500 -> exact
    const int m = row / NTOT;
    const int i = row % NTOT;
    const float* src = (m == 0) ? a : (m == 1) ? v : l;

    int spk = 0;
    if (m == 2) {
        const int b  = i / LD;
        const int li = i % LD;
        // qmask layout (L, B, 2); argmax ties -> index 0
        const float q0 = qmask[(li * BD + b) * 2 + 0];
        const float q1 = qmask[(li * BD + b) * 2 + 1];
        spk = (q1 > q0) ? 1 : 0;
    }

    float vals[8];
    float ss = 0.0f;
#pragma unroll
    for (int t = 0; t < 8; ++t) {
        const int d = lane + t * 32;
        float x = src[i * DIM + d];
        if (m == 2) x += spk_emb[spk * DIM + d];
        vals[t] = x;
        ss += x * x;
    }
#pragma unroll
    for (int o = 16; o > 0; o >>= 1) ss += __shfl_xor(ss, o, 32);
    const float rn = rsqrtf(ss);
#pragma unroll
    for (int t = 0; t < 8; ++t) {
        const int d = lane + t * 32;
        X  [row * DIM + d] = vals[t];
        X16[row * DIM + d] = (_Float16)vals[t];
        NRM[row * DIM + d] = vals[t] * rn;
    }
}

// ---------------------------------------------------------------------------
// convw: pack 5 weight matrices (W0, Wconvs[0..3]) into fragment-ordered f16
// WF[w][colTile(16)][k0(8)][lane(32)][j(16)]  (ISA 7.12.2 16-bit B 32x16)
// ---------------------------------------------------------------------------
__global__ __launch_bounds__(256)
void convw_kernel(const float* __restrict__ W0, const float* __restrict__ Wconvs,
                  _Float16* __restrict__ WF)
{
    const int gid = blockIdx.x * 256 + threadIdx.x;   // grid covers 5*WFRAG
    const int w   = gid / WFRAG;
    const int rem = gid % WFRAG;
    const int ct   = rem >> 12;        // colTile
    const int k0i  = (rem >> 9) & 7;   // k block
    const int lane = (rem >> 4) & 31;
    const int j    = rem & 15;
    const int k = k0i * 32 + (j >> 3) * 16 + (lane >> 4) * 8 + (j & 7);
    const int n = ct * 16 + (lane & 15);
    const float* Wsrc = (w == 0) ? W0 : (Wconvs + (w - 1) * DIM * DIM);
    WF[gid] = (_Float16)Wsrc[k * DIM + n];
}

// ---------------------------------------------------------------------------
// sim: per (modality, dialogue) 40x40 angular similarity block
// ---------------------------------------------------------------------------
__global__ __launch_bounds__(256)
void sim_kernel(const float* __restrict__ NRM, float* __restrict__ S)
{
    __shared__ float Xs[LD * 257];
    const int mb = blockIdx.x;
    const int m  = mb / BD;
    const int b  = mb % BD;
    const int base = (m * NTOT + b * LD) * DIM;

    for (int idx = threadIdx.x; idx < LD * DIM; idx += 256) {
        const int r = idx >> 8, c = idx & 255;
        Xs[r * 257 + c] = NRM[base + idx];
    }
    __syncthreads();

    const float INVPI = 0.31830988618379067f;
    for (int e = threadIdx.x; e < LD * LD; e += 256) {
        const int i = e / LD, j = e % LD;
        float dot = 0.0f;
#pragma unroll 8
        for (int k = 0; k < DIM; ++k) dot += Xs[i * 257 + k] * Xs[j * 257 + k];
        dot *= 0.99999f;
        dot = fminf(fmaxf(dot, -1.0f), 1.0f);
        S[mb * (LD * LD) + e] = 1.0f - acosf(dot) * INVPI;
    }
}

// ---------------------------------------------------------------------------
// cross: per-utterance cross-modal angular sims c01,c02,c12 (MODAL_WEIGHT=1)
// ---------------------------------------------------------------------------
__global__ __launch_bounds__(256)
void cross_kernel(const float* __restrict__ NRM, float* __restrict__ CROSS)
{
    const int i = blockIdx.x * blockDim.x + threadIdx.x;
    if (i >= NTOT) return;
    const float* r0 = NRM + (0 * NTOT + i) * DIM;
    const float* r1 = NRM + (1 * NTOT + i) * DIM;
    const float* r2 = NRM + (2 * NTOT + i) * DIM;
    float d01 = 0.f, d02 = 0.f, d12 = 0.f;
    for (int k = 0; k < DIM; ++k) {
        const float x0 = r0[k], x1 = r1[k], x2 = r2[k];
        d01 += x0 * x1; d02 += x0 * x2; d12 += x1 * x2;
    }
    const float INVPI = 0.31830988618379067f;
    d01 = fminf(fmaxf(d01 * 0.99999f, -1.f), 1.f);
    d02 = fminf(fmaxf(d02 * 0.99999f, -1.f), 1.f);
    d12 = fminf(fmaxf(d12 * 0.99999f, -1.f), 1.f);
    CROSS[i * 3 + 0] = 1.0f - acosf(d01) * INVPI;
    CROSS[i * 3 + 1] = 1.0f - acosf(d02) * INVPI;
    CROSS[i * 3 + 2] = 1.0f - acosf(d12) * INVPI;
}

// ---------------------------------------------------------------------------
// degree: d^{-1/2} per row (40 block entries + 2 cross entries)
// ---------------------------------------------------------------------------
__global__ __launch_bounds__(256)
void degree_kernel(const float* __restrict__ S, const float* __restrict__ CROSS,
                   float* __restrict__ DINV)
{
    const int row = blockIdx.x * blockDim.x + threadIdx.x;
    if (row >= ROWS3) return;
    const int m = row / NTOT, i = row % NTOT, b = i / LD, li = i % LD;
    const float* srow = S + (m * BD + b) * (LD * LD) + li * LD;
    float d = 0.0f;
#pragma unroll
    for (int j = 0; j < LD; ++j) d += srow[j];
    if (m == 0)      d += CROSS[i * 3 + 0] + CROSS[i * 3 + 1];
    else if (m == 1) d += CROSS[i * 3 + 0] + CROSS[i * 3 + 2];
    else             d += CROSS[i * 3 + 1] + CROSS[i * 3 + 2];
    DINV[row] = rsqrtf(d);
}

// ---------------------------------------------------------------------------
// support: SUP = 0.9 * (Adjnorm @ H) + 0.1 * H0   (42 nnz per row)
// one block per row; writes f32 (for theta blend) and f16 (GEMM A operand)
// ---------------------------------------------------------------------------
__global__ __launch_bounds__(256)
void support_kernel(const float* __restrict__ S, const float* __restrict__ CROSS,
                    const float* __restrict__ DINV,
                    const float* __restrict__ H, const float* __restrict__ H0,
                    float* __restrict__ SUP, _Float16* __restrict__ SUP16)
{
    __shared__ float w[LD + 2];
    __shared__ int   nidx[2];
    const int row = blockIdx.x;                     // 0..11999
    const int m = row / NTOT, i = row % NTOT, b = i / LD, li = i % LD;
    const float di = DINV[row];

    if (threadIdx.x < LD) {
        const int j = threadIdx.x;
        w[j] = S[(m * BD + b) * (LD * LD) + li * LD + j] * di
             * DINV[m * NTOT + b * LD + j];
    } else if (threadIdx.x < LD + 2) {
        const int t  = threadIdx.x - LD;            // 0 or 1
        const int n0 = (m == 0) ? 1 : 0;
        const int n1 = (m == 2) ? 1 : 2;
        const int n  = (t == 0) ? n0 : n1;
        const int lo = (m < n) ? m : n;
        const int hi = (m < n) ? n : m;
        const int ci = (lo == 0) ? ((hi == 1) ? 0 : 1) : 2;  // (0,1)(0,2)(1,2)
        w[LD + t]  = CROSS[i * 3 + ci] * di * DINV[n * NTOT + i];
        nidx[t]    = n * NTOT + i;
    }
    __syncthreads();

    const int d = threadIdx.x;                      // 256 dims
    const int g0 = m * NTOT + b * LD;
    float acc = 0.0f;
#pragma unroll
    for (int j = 0; j < LD; ++j) acc += w[j] * H[(g0 + j) * DIM + d];
    acc += w[LD + 0] * H[nidx[0] * DIM + d];
    acc += w[LD + 1] * H[nidx[1] * DIM + d];
    const float s = 0.9f * acc + 0.1f * H0[row * DIM + d];
    SUP  [row * DIM + d] = s;
    SUP16[row * DIM + d] = (_Float16)s;
}

// ---------------------------------------------------------------------------
// WMMA GEMM: C[12000x256] = A16[12000x256] @ W[256x256], f16 in / f32 acc
// 512 threads = 16 waves; wave w owns output tile (blockIdx*16, w*16)
// A panel staged as f16 in LDS (stride 264 f16 -> bank spread); fragments:
//   A: two ds_load_b128 (two contiguous 8-f16 runs per ISA 16-bit A 16x32)
//   B: one contiguous 32B/lane load from fragment-ordered WF (global, L2)
// mode 0: out = relu(acc + bias)        -> writes Out0 and Out1 (H0 and H)
// mode 1: out = relu(th*acc + (1-th)*Sup) -> writes Out0 (H)
// ---------------------------------------------------------------------------
__global__ __launch_bounds__(512)
void wmma_gemm_kernel(const _Float16* __restrict__ A16,
                      const _Float16* __restrict__ WF,
                      const float* __restrict__ bias,
                      const float* __restrict__ Sup,
                      float* __restrict__ Out0, float* __restrict__ Out1,
                      int mode, float theta)
{
    __shared__ __attribute__((aligned(32))) _Float16 AsH[16 * 264];
    const int rowBase = blockIdx.x * 16;

    // stage A panel: 16 rows x 256 f16 = 2048 dwords, coalesced
    {
        const unsigned int* Ag = (const unsigned int*)(A16 + rowBase * DIM);
        unsigned int* AsW = (unsigned int*)AsH;
        for (int idx = threadIdx.x; idx < 16 * (DIM / 2); idx += 512) {
            const int r = idx >> 7, c2 = idx & 127;       // dword col
            AsW[r * 132 + c2] = Ag[idx];                  // row stride 132 dw
        }
    }
    __syncthreads();

    const int wave = threadIdx.x >> 5;
    const int lane = threadIdx.x & 31;
    const int colBase = wave * 16;
    const int r    = lane & 15;
    const int hi   = lane >> 4;
    const int kb8  = hi * 8;     // A-fragment K base within 32-wide k block

    v8f acc = {};
#pragma unroll
    for (int k0i = 0; k0i < 8; ++k0i) {
        const _Float16* arow = &AsH[r * 264 + k0i * 32 + kb8];
        const v8h alo = *(const v8h*)(arow);        // K = kb8 + 0..7
        const v8h ahi = *(const v8h*)(arow + 16);   // K = 16 + kb8 + 0..7
        const v16h af = __builtin_shufflevector(alo, ahi,
                          0, 1, 2, 3, 4, 5, 6, 7, 8, 9, 10, 11, 12, 13, 14, 15);
        const v16h bf = *(const v16h*)(WF + (((wave * 8) + k0i) * 32 + lane) * 16);
        acc = __builtin_amdgcn_wmma_f32_16x16x32_f16(
                  false, af, false, bf, (short)0, acc, false, false);
    }

    // C/D layout: element e -> M = e + 8*hi, N = lane&15
    const int mOff = hi * 8;
    const int gcol = colBase + (lane & 15);
#pragma unroll
    for (int e = 0; e < 8; ++e) {
        const int grow = rowBase + mOff + e;
        float v = acc[e];
        if (mode == 0) {
            v = fmaxf(v + bias[gcol], 0.0f);
            Out0[grow * DIM + gcol] = v;
            Out1[grow * DIM + gcol] = v;
        } else {
            const float s = Sup[grow * DIM + gcol];
            v = fmaxf(theta * v + (1.0f - theta) * s, 0.0f);
            Out0[grow * DIM + gcol] = v;
        }
    }
}

// ---------------------------------------------------------------------------
// assemble: out[i, m*512 + d] = (d<256 ? X : H)[m*N + i, d&255]
// ---------------------------------------------------------------------------
__global__ __launch_bounds__(256)
void assemble_kernel(const float* __restrict__ X, const float* __restrict__ H,
                     float* __restrict__ out)
{
    const int gid = blockIdx.x * blockDim.x + threadIdx.x;
    if (gid >= NTOT * 1536) return;
    const int i = gid / 1536, o = gid % 1536;
    const int m = o / 512,    d = o % 512;
    const float* src = (d < 256) ? X : H;
    out[gid] = src[(m * NTOT + i) * DIM + (d & 255)];
}

// ---------------------------------------------------------------------------
// launch
// ---------------------------------------------------------------------------
extern "C" void kernel_launch(void* const* d_in, const int* in_sizes, int n_in,
                              void* d_out, int out_size, void* d_ws, size_t ws_size,
                              hipStream_t stream)
{
    const float* a       = (const float*)d_in[0];
    const float* v       = (const float*)d_in[1];
    const float* l       = (const float*)d_in[2];
    const float* qmask   = (const float*)d_in[3];
    const float* spk_emb = (const float*)d_in[4];
    const float* W0      = (const float*)d_in[5];
    const float* b0      = (const float*)d_in[6];
    const float* Wconvs  = (const float*)d_in[7];
    float* out = (float*)d_out;

    // workspace layout (floats): ~77 MB total, all L2-resident
    float* ws    = (float*)d_ws;
    float* X     = ws;                           // 3N*256
    float* NRM   = X     + ROWS3 * DIM;          // 3N*256
    float* H0    = NRM   + ROWS3 * DIM;          // 3N*256
    float* H     = H0    + ROWS3 * DIM;          // 3N*256
    float* SUP   = H     + ROWS3 * DIM;          // 3N*256
    float* S     = SUP   + ROWS3 * DIM;          // 3*BD*1600
    float* CROSS = S     + 3 * BD * LD * LD;     // 3*N
    float* DINV  = CROSS + 3 * NTOT;             // 3N
    _Float16* X16   = (_Float16*)(DINV + ROWS3);       // 3N*256 f16
    _Float16* SUP16 = X16 + ROWS3 * DIM;               // 3N*256 f16
    _Float16* WF    = SUP16 + ROWS3 * DIM;             // 5*65536 f16

    prep_kernel  <<<ROWS3 / 8, 256, 0, stream>>>(a, v, l, qmask, spk_emb,
                                                 X, X16, NRM);
    convw_kernel <<<5 * WFRAG / 256, 256, 0, stream>>>(W0, Wconvs, WF);
    sim_kernel   <<<3 * BD,    256, 0, stream>>>(NRM, S);
    cross_kernel <<<(NTOT + 255) / 256,  256, 0, stream>>>(NRM, CROSS);
    degree_kernel<<<(ROWS3 + 255) / 256, 256, 0, stream>>>(S, CROSS, DINV);

    // h0 = relu(X @ W0 + b0); also initializes H = h0
    wmma_gemm_kernel<<<ROWS3 / 16, 512, 0, stream>>>(X16, WF, b0, SUP,
                                                     H0, H, 0, 0.0f);

    const float LAMDA = 0.5f, thetas[4] = {
        logf(LAMDA / 1.0f + 1.0f), logf(LAMDA / 2.0f + 1.0f),
        logf(LAMDA / 3.0f + 1.0f), logf(LAMDA / 4.0f + 1.0f) };

    for (int it = 0; it < 4; ++it) {
        support_kernel<<<ROWS3, 256, 0, stream>>>(S, CROSS, DINV, H, H0,
                                                  SUP, SUP16);
        wmma_gemm_kernel<<<ROWS3 / 16, 512, 0, stream>>>(
            SUP16, WF + (it + 1) * WFRAG, b0, SUP, H, H, 1, thetas[it]);
    }

    assemble_kernel<<<(NTOT * 1536 + 255) / 256, 256, 0, stream>>>(X, H, out);
    (void)in_sizes; (void)n_in; (void)out_size; (void)ws_size;
}
